// GraphBlock_18245021073648
// MI455X (gfx1250) — matrix operand order repaired and checked by reading, not compiled, not run on hardware
//
#include <hip/hip_runtime.h>
#include <math.h>
#include <stdint.h>

// ---------------- CDNA5 wave32 WMMA / TDM types ----------------
typedef __attribute__((ext_vector_type(16))) __bf16 v16bf;
typedef __attribute__((ext_vector_type(8)))  __bf16 v8bf;
typedef __attribute__((ext_vector_type(8)))  float  v8f;
typedef __attribute__((ext_vector_type(4))) unsigned int u32x4;
typedef __attribute__((ext_vector_type(8)))  int i32x8;
typedef __attribute__((ext_vector_type(4)))  int i32x4;

#define N_NODES 512
#define BATCH   8
#define SEQ     512
#define MROWS   (BATCH * SEQ)     // 4096 rows of the node-diffusion GEMM
#define ALPHA   0.05f
#define LN_EPS  1e-5f

// =====================================================================
// Kernel 1: learned adjacency a[v,w]; emits B-operand directly in the
// per-lane WMMA fragment layout:
//   Bpack[nTile][kStep][lane][j] = B[kStep*32 + (lane>>4)*16 + j, nTile*16 + (lane&15)]
// where B[k][n] = a[n][k]  (so GEMM computes z @ a^T).
// One block per row v, 512 threads (one per w).
// =====================================================================
__global__ void build_adj(const float* __restrict__ nodevec,
                          const float* __restrict__ nodeveck,
                          __bf16* __restrict__ Bpack)
{
    const int v = blockIdx.x;
    const int w = threadIdx.x;
    __shared__ float nvv[10];
    __shared__ float red[N_NODES];
    __shared__ int   redi[N_NODES];
    __shared__ float s2[N_NODES];

    if (w < 10) nvv[w] = nodevec[v * 10 + w] * nodeveck[w];
    __syncthreads();

    float dot = 0.f;
#pragma unroll
    for (int d = 0; d < 10; ++d)
        dot += nvv[d] * nodevec[w * 10 + d] * nodeveck[d];
    float sc = dot > 0.f ? dot : 0.f;

    // --- row softmax: max then sum ---
    red[w] = sc; __syncthreads();
    for (int st = 256; st; st >>= 1) {
        if (w < st) red[w] = fmaxf(red[w], red[w + st]);
        __syncthreads();
    }
    float mx = red[0]; __syncthreads();
    float e = __expf(sc - mx);
    red[w] = e; __syncthreads();
    for (int st = 256; st; st >>= 1) {
        if (w < st) red[w] += red[w + st];
        __syncthreads();
    }
    float soft = e / red[0];
    __syncthreads();

    // --- top-4 mask: 4 argmax passes, winners marked -1 in s2 ---
    s2[w] = soft; __syncthreads();
    for (int t = 0; t < 4; ++t) {
        red[w] = s2[w]; redi[w] = w; __syncthreads();
        for (int st = 256; st; st >>= 1) {
            if (w < st && red[w + st] > red[w]) { red[w] = red[w + st]; redi[w] = redi[w + st]; }
            __syncthreads();
        }
        if (w == 0) s2[redi[0]] = -1.f;
        __syncthreads();
    }
    float masked = (s2[w] < 0.f) ? soft : 0.f;
    if (w == v) masked += 1.f;               // self loop

    red[w] = masked; __syncthreads();
    for (int st = 256; st; st >>= 1) {
        if (w < st) red[w] += red[w + st];
        __syncthreads();
    }
    float aval = masked / red[0];

    // scatter into packed WMMA-B fragment layout: k = w, n = v
    const int tn   = v >> 4;
    const int lm   = v & 15;
    const int ks   = w >> 5;
    const int half = (w >> 4) & 1;
    const int j    = w & 15;
    const int lanep = half * 16 + lm;
    Bpack[(((size_t)tn * 16 + ks) * 32 + lanep) * 16 + j] = (__bf16)aval;
}

// =====================================================================
// Kernel 2: f32 -> bf16 conversion of x (GEMM A-operand, row-major)
// =====================================================================
__global__ void cvt_bf16(const float* __restrict__ in, __bf16* __restrict__ o)
{
    size_t i = (size_t)blockIdx.x * blockDim.x + threadIdx.x;
    const size_t stride = (size_t)gridDim.x * blockDim.x;
    const size_t total = (size_t)MROWS * N_NODES;
    for (; i < total; i += stride) o[i] = (__bf16)in[i];
}

// =====================================================================
// Kernel 3: node-diffusion GEMM  z' = ALPHA*x + (1-ALPHA) * Z @ a^T
// Block = 8 waves = one 16-row A stripe x all 512 cols.
//  - A stripe (16x512 bf16) staged into LDS once per block by the Tensor
//    Data Mover (2D tile, 16B pad per 1024B row -> conflict-free ds_load).
//  - Each wave owns 4 N-tiles (16x64 output), 4 accumulators:
//    per k-step 2 LDS loads (A) + 4x32B packed B loads + 4 WMMAs.
// =====================================================================
template<bool WF, bool WB>
__global__ void __launch_bounds__(256)
diffuse_gemm(const __bf16* __restrict__ Zbf,
             const __bf16* __restrict__ Bpack,
             const float*  __restrict__ x,
             float*  __restrict__ outF,
             __bf16* __restrict__ outBf)
{
    constexpr int LDSROW = 520;               // 512 elems + 8 pad elems (16B)
    __shared__ __bf16 sA[16 * 528];

    const int tm   = blockIdx.x;              // 256 blocks, one 16-row stripe
    const int wv   = threadIdx.x >> 5;        // wave -> group of 4 N-tiles
    const int lane = threadIdx.x & 31;
    const int half = lane >> 4, lm = lane & 15;
    const int m0   = tm * 16;

    if (wv == 0) {
        // ---- TDM descriptor: 2D tile 512(x) x 16(y) of 2-byte elems ----
        unsigned long long ga = (unsigned long long)(uintptr_t)(Zbf + (size_t)m0 * N_NODES);
        unsigned ldsa = (unsigned)(uintptr_t)(&sA[0]);
        u32x4 g0;
        g0[0] = 1u;                                          // count=1 (valid)
        g0[1] = ldsa;                                        // lds_addr
        g0[2] = (unsigned)(ga & 0xFFFFFFFFu);                // global_addr lo
        g0[3] = (unsigned)((ga >> 32) & 0x01FFFFFFu) | (2u << 30); // hi + type=2
        i32x8 g1;
        // data_size=2B | pad_enable | pad_interval=256dw(1024B) | pad_amount=4dw(16B)
        g1[0] = (1 << 16) | (1 << 20) | (7 << 22) | (3 << 25);
        g1[1] = (int)(N_NODES << 16);            // tensor_dim0 = 512 (lo16)
        g1[2] = (int)((MROWS & 0xFFFF) << 16);   // tensor_dim1 = 4096 (lo16)
        g1[3] = (int)(N_NODES << 16);            // tile_dim0 = 512
        g1[4] = 16;                              // tile_dim1 = 16, tile_dim2 = 0
        g1[5] = N_NODES;                         // tensor_dim0_stride = 512
        g1[6] = 0;
        g1[7] = 0;
        i32x4 gz4 = {0, 0, 0, 0};
        i32x8 gz8 = {0, 0, 0, 0, 0, 0, 0, 0};
        __builtin_amdgcn_tensor_load_to_lds(g0, g1, gz4, gz4, gz8, 0);
        __builtin_amdgcn_s_wait_tensorcnt(0);
    }
    __syncthreads();

    v8f acc[4] = {};
    const __bf16* aRow = &sA[lm * LDSROW];
    const __bf16* bBase = Bpack + (((size_t)(wv * 4) * 16) * 32 + lane) * 16;
    constexpr size_t TILE_STRIDE = (size_t)16 * 32 * 16;   // one N-tile of packed B
    constexpr size_t KS_STRIDE   = (size_t)32 * 16;        // one k-step inside a tile

    for (int ks = 0; ks < 16; ++ks) {
        const int kb = ks * 32;
        v8bf alo = *(const v8bf*)(aRow + kb + half * 16);
        v8bf ahi = *(const v8bf*)(aRow + kb + half * 16 + 8);
        v16bf a;
#pragma unroll
        for (int j = 0; j < 8; ++j) { a[j] = alo[j]; a[8 + j] = ahi[j]; }

        const __bf16* bp = bBase + (size_t)ks * KS_STRIDE;
        v16bf b0 = *(const v16bf*)(bp);
        v16bf b1 = *(const v16bf*)(bp + TILE_STRIDE);
        v16bf b2 = *(const v16bf*)(bp + 2 * TILE_STRIDE);
        v16bf b3 = *(const v16bf*)(bp + 3 * TILE_STRIDE);

        acc[0] = __builtin_amdgcn_wmma_f32_16x16x32_bf16(false, a, false, b0, (short)0, acc[0], false, false);
        acc[1] = __builtin_amdgcn_wmma_f32_16x16x32_bf16(false, a, false, b1, (short)0, acc[1], false, false);
        acc[2] = __builtin_amdgcn_wmma_f32_16x16x32_bf16(false, a, false, b2, (short)0, acc[2], false, false);
        acc[3] = __builtin_amdgcn_wmma_f32_16x16x32_bf16(false, a, false, b3, (short)0, acc[3], false, false);
    }

    // C/D layout: lane -> col (lane&15), VGPR r -> row m0 + (lane>>4)*8 + r
#pragma unroll
    for (int t = 0; t < 4; ++t) {
#pragma unroll
        for (int r = 0; r < 8; ++r) {
            const int m = m0 + half * 8 + r;
            const size_t id = (size_t)m * N_NODES + (wv * 64 + t * 16 + lm);
            float z = ALPHA * x[id] + (1.0f - ALPHA) * acc[t][r];
            if (WF) outF[id]  = z;
            if (WB) outBf[id] = (__bf16)z;
        }
    }
}

// =====================================================================
// Kernel 4: fused pointwise MLP (rank-1 collapsed) + GELU + residual + LN
// One block per (b,l) row, 512 threads (one per node).
// =====================================================================
__global__ void fused_epilogue(const float* __restrict__ x,
                               const __bf16* __restrict__ z1bf,
                               const float* __restrict__ z2,
                               const float* __restrict__ w_mlp,
                               const float* __restrict__ b_mlp,
                               const float* __restrict__ w_start,
                               const float* __restrict__ b_start,
                               const float* __restrict__ w_end,
                               const float* __restrict__ b_end,
                               const float* __restrict__ gamma,
                               const float* __restrict__ beta,
                               float* __restrict__ out)
{
    __shared__ float u0[16], u1[16], u2[16], c0[16], we[16];
    __shared__ float ps[16], pq[16];
    __shared__ float smu, srstd;
    const int n = threadIdx.x;
    const int m = blockIdx.x;

    if (n < 16) {           // fold w_mlp . (w_start, b_start) once per block
        const int o = n;
        float a0 = 0.f, a1 = 0.f, a2 = 0.f, cc = b_mlp[o];
#pragma unroll
        for (int c = 0; c < 16; ++c) {
            float wsv = w_start[c], bsv = b_start[c];
            float m0v = w_mlp[o * 48 + c];
            float m1v = w_mlp[o * 48 + 16 + c];
            float m2v = w_mlp[o * 48 + 32 + c];
            a0 += m0v * wsv; a1 += m1v * wsv; a2 += m2v * wsv;
            cc += (m0v + m1v + m2v) * bsv;
        }
        u0[o] = a0; u1[o] = a1; u2[o] = a2; c0[o] = cc; we[o] = w_end[o];
    }
    __syncthreads();

    const size_t id = (size_t)m * N_NODES + n;
    float zz0 = x[id];
    float zz1 = (float)z1bf[id];
    float zz2 = z2[id];
    float acc = b_end[0];
#pragma unroll
    for (int o = 0; o < 16; ++o) {
        float t = u0[o] * zz0 + u1[o] * zz1 + u2[o] * zz2 + c0[o];
        float g = 0.5f * t * (1.0f + erff(t * 0.70710678118654752f)); // exact GELU
        acc += we[o] * g;
    }
    float y = zz0 + acc;

    // LayerNorm over node dim: wave32 shuffle + cross-wave shared reduce
    float s = y, q = y * y;
    for (int off = 16; off; off >>= 1) {
        s += __shfl_down(s, off);
        q += __shfl_down(q, off);
    }
    const int wid = n >> 5, lane = n & 31;
    if (lane == 0) { ps[wid] = s; pq[wid] = q; }
    __syncthreads();
    if (wid == 0) {
        float s2v = (lane < 16) ? ps[lane] : 0.f;
        float q2v = (lane < 16) ? pq[lane] : 0.f;
        for (int off = 8; off; off >>= 1) {
            s2v += __shfl_down(s2v, off);
            q2v += __shfl_down(q2v, off);
        }
        if (lane == 0) {
            float mu  = s2v * (1.0f / N_NODES);
            float var = q2v * (1.0f / N_NODES) - mu * mu;
            smu   = mu;
            srstd = rsqrtf(var + LN_EPS);
        }
    }
    __syncthreads();
    out[id] = (y - smu) * srstd * gamma[n] + beta[n];
}

// =====================================================================
extern "C" void kernel_launch(void* const* d_in, const int* in_sizes, int n_in,
                              void* d_out, int out_size, void* d_ws, size_t ws_size,
                              hipStream_t stream)
{
    (void)in_sizes; (void)n_in; (void)out_size; (void)ws_size;

    const float* x        = (const float*)d_in[0];
    const float* nodevec  = (const float*)d_in[1];
    const float* nodeveck = (const float*)d_in[2];
    const float* w_start  = (const float*)d_in[3];
    const float* b_start  = (const float*)d_in[4];
    const float* w_mlp    = (const float*)d_in[5];
    const float* b_mlp    = (const float*)d_in[6];
    const float* w_end    = (const float*)d_in[7];
    const float* b_end    = (const float*)d_in[8];
    const float* gamma    = (const float*)d_in[9];
    const float* beta     = (const float*)d_in[10];
    float* out = (float*)d_out;

    // workspace layout (16.5 MB total)
    char* ws = (char*)d_ws;
    const size_t szB  = (size_t)N_NODES * N_NODES * sizeof(__bf16);   // 0.5 MB packed B
    const size_t szZb = (size_t)MROWS * N_NODES * sizeof(__bf16);     // 4 MB
    __bf16* Bpack = (__bf16*)(ws);
    __bf16* z0bf  = (__bf16*)(ws + szB);
    __bf16* z1bf  = (__bf16*)(ws + szB + szZb);
    float*  z2f   = (float*) (ws + szB + 2 * szZb);

    // 1. learned adjacency (top-4 + self loop, row-normalized) -> packed WMMA-B
    build_adj<<<N_NODES, N_NODES, 0, stream>>>(nodevec, nodeveck, Bpack);
    // 2. bf16 copy of x for GEMM A-operand
    cvt_bf16<<<1024, 256, 0, stream>>>(x, z0bf);
    // 3. diffusion step 1: z1 = a*x + (1-a) z0 @ A^T   (bf16 out)
    diffuse_gemm<false, true><<<256, 256, 0, stream>>>(z0bf, Bpack, x, nullptr, z1bf);
    // 4. diffusion step 2: z2 = a*x + (1-a) z1 @ A^T   (f32 out)
    diffuse_gemm<true, false><<<256, 256, 0, stream>>>(z1bf, Bpack, x, z2f, nullptr);
    // 5. fused collapsed-MLP + GELU + residual + LayerNorm
    fused_epilogue<<<MROWS, N_NODES, 0, stream>>>(x, z1bf, z2f, w_mlp, b_mlp,
                                                  w_start, b_start, w_end, b_end,
                                                  gamma, beta, out);
}
